// SimpleTransformer_19069654794358
// MI455X (gfx1250) — matrix operand order repaired and checked by reading, not compiled, not run on hardware
//
#include <hip/hip_runtime.h>

#define kB 2
#define kS 1024
#define kE 1024
#define kH 16
#define kD 64
#define kV 32000
#define kF 512

typedef __attribute__((ext_vector_type(16))) __bf16 v16bf;
typedef __attribute__((ext_vector_type(8)))  __bf16 bf16x8;
typedef __attribute__((ext_vector_type(8)))  float  v8f;

__device__ inline v8f zero8() {
  v8f z;
#pragma unroll
  for (int i = 0; i < 8; ++i) z[i] = 0.0f;
  return z;
}

// ===========================================================================
// Fragment loaders
// ===========================================================================
// A fragment 16x32 (MxK) from bf16 row-major: lane needs k = [kh..kh+15]
// (kh = 16*(lane/16)) -> two 16B loads.
__device__ inline v16bf load_a_frag(const __bf16* __restrict__ A, int lda, int lane) {
  const int m  = lane & 15;
  const int kh = (lane >> 4) << 4;
  const __bf16* row = A + (size_t)m * lda + kh;
  const bf16x8 lo = *(const bf16x8*)(row);
  const bf16x8 hi = *(const bf16x8*)(row + 8);
  return __builtin_shufflevector(lo, hi, 0, 1, 2, 3, 4, 5, 6, 7,
                                         8, 9, 10, 11, 12, 13, 14, 15);
}

// B fragment 32x16 from packed tile (512 bf16): lane's 16 values are the 32
// contiguous bytes at tile + lane*32 (pack_b_kernel emits this order).
__device__ inline v16bf load_b_packed(const __bf16* __restrict__ T, int lane) {
  const __bf16* p = T + lane * 16;
  const bf16x8 lo = *(const bf16x8*)(p);
  const bf16x8 hi = *(const bf16x8*)(p + 8);
  return __builtin_shufflevector(lo, hi, 0, 1, 2, 3, 4, 5, 6, 7,
                                         8, 9, 10, 11, 12, 13, 14, 15);
}

// ---- f32-source loaders used by the attention kernel (contiguous per lane) ----
__device__ inline v16bf load_a_bf32(const float* __restrict__ A, int lda, int lane) {
  const int m  = lane & 15;
  const int kh = (lane >> 4) << 4;
  const float* row = A + (size_t)m * lda;
  v16bf r;
#pragma unroll
  for (int v = 0; v < 8; ++v) {
    const int k = kh + (v << 1);
    r[2 * v]     = (__bf16)row[k];
    r[2 * v + 1] = (__bf16)row[k + 1];
  }
  return r;
}

__device__ inline v16bf load_bT_bf(const float* __restrict__ Kp, int kvb, int koff, int lane) {
  const int n  = lane & 15;
  const int kb = (lane >> 4) << 4;
  const float* row = Kp + (size_t)(kvb + n) * kD + koff + kb;
  v16bf r;
#pragma unroll
  for (int v = 0; v < 8; ++v) {
    r[2 * v]     = (__bf16)row[2 * v];
    r[2 * v + 1] = (__bf16)row[2 * v + 1];
  }
  return r;
}

__device__ inline v16bf load_bv_bf(const float* __restrict__ Vp, int kvb, int coff,
                                   int Slim, int lane) {
  const int n  = lane & 15;
  const int kb = (lane >> 4) << 4;
  v16bf r;
#pragma unroll
  for (int v = 0; v < 8; ++v) {
    int k0 = kvb + kb + 2 * v;
    int k1 = k0 + 1;
    if (k0 >= Slim) k0 = Slim - 1;
    if (k1 >= Slim) k1 = Slim - 1;
    r[2 * v]     = (__bf16)Vp[(size_t)k0 * kD + coff + n];
    r[2 * v + 1] = (__bf16)Vp[(size_t)k1 * kD + coff + n];
  }
  return r;
}

__device__ inline v16bf load_ap_bf(const float (*p)[32], int lane) {
  const int m  = lane & 15;
  const int kh = (lane >> 4) << 4;
  v16bf r;
#pragma unroll
  for (int v = 0; v < 8; ++v) {
    const int k = kh + (v << 1);
    r[2 * v]     = (__bf16)p[m][k];
    r[2 * v + 1] = (__bf16)p[m][k + 1];
  }
  return r;
}

// ===========================================================================
// Operand packing kernels (f32 -> bf16)
// ===========================================================================
__global__ __launch_bounds__(256) void pack_a_kernel(const float* __restrict__ in,
                                                     __bf16* __restrict__ outp, int n) {
  const int i = (blockIdx.x * 256 + threadIdx.x) * 4;
  if (i >= n) return;
  const float4 v = *(const float4*)(in + i);
  outp[i + 0] = (__bf16)v.x;
  outp[i + 1] = (__bf16)v.y;
  outp[i + 2] = (__bf16)v.z;
  outp[i + 3] = (__bf16)v.w;
}

// Pack W[K,N] f32 -> bf16 tiles of 512 elements. Tile index = nt*(K/32)+kt so the
// GEMM K-loop streams tiles contiguously. Within a tile, element order is
// (khalf*16 + n)*16 + klocal, i.e. exactly the B-fragment lane order.
__global__ __launch_bounds__(256) void pack_b_kernel(const float* __restrict__ W,
                                                     __bf16* __restrict__ outp,
                                                     int K, int N,
                                                     long long inZ, long long outZ) {
  const float* w = W + (long long)blockIdx.z * inZ;
  __bf16* o = outp + (long long)blockIdx.z * outZ + (size_t)blockIdx.x * 512;
  const int ktiles = K >> 5;
  const int nt = blockIdx.x / ktiles;
  const int kt = blockIdx.x % ktiles;
  for (int e = threadIdx.x; e < 512; e += 256) {
    const int khalf = e >> 8;
    const int nl = (e >> 4) & 15;
    const int kl = e & 15;
    const int k = kt * 32 + khalf * 16 + kl;
    const int n = nt * 16 + nl;
    o[e] = (__bf16)w[(size_t)k * N + n];
  }
}

// ===========================================================================
// WMMA GEMM on packed bf16 operands: C[M,N] = A[M,K] @ W[K,N] + bias (opt ReLU).
// Block = 8 waves; wave w: rows [y*256 + w*32, +32), cols [x*64, +64).
// All six fragments of a K-step are loaded before any WMMA so the scheduler can
// issue one load clause and stagger partial loadcnt waits across the 8 WMMAs.
// grid.z batching: A += (z/Hmod)*aZ ; Wp += (z%Hmod)*wZ ; bias += (z%Hmod)*bZ ; C += z*cZ.
// ===========================================================================
__global__ __launch_bounds__(256) void gemm_bf_kernel(
    const __bf16* __restrict__ A, const __bf16* __restrict__ Wp,
    const float* __restrict__ bias, float* __restrict__ C,
    int M, int N, int K, int lda, int ldc,
    long long aZ, long long wZ, long long bZ, long long cZ, int Hmod, int relu)
{
  const int z = blockIdx.z;
  A    += (long long)(z / Hmod) * aZ;
  Wp   += (long long)(z % Hmod) * wZ;
  bias += (long long)(z % Hmod) * bZ;
  C    += (long long)z * cZ;
  const int lane = threadIdx.x & 31;
  const int wave = threadIdx.x >> 5;
  const int row0 = blockIdx.y * 256 + wave * 32;
  const int col0 = blockIdx.x * 64;
  if (row0 >= M) return;
  const int ktiles = K >> 5;

  v8f acc[2][4];
#pragma unroll
  for (int r = 0; r < 2; ++r)
#pragma unroll
    for (int t = 0; t < 4; ++t) acc[r][t] = zero8();

  const __bf16* arow0 = A + (size_t)row0 * lda;
  const __bf16* arow1 = arow0 + (size_t)16 * lda;
  const __bf16* wt0 = Wp + ((size_t)((col0 >> 4) + 0) * ktiles) * 512;
  const __bf16* wt1 = Wp + ((size_t)((col0 >> 4) + 1) * ktiles) * 512;
  const __bf16* wt2 = Wp + ((size_t)((col0 >> 4) + 2) * ktiles) * 512;
  const __bf16* wt3 = Wp + ((size_t)((col0 >> 4) + 3) * ktiles) * 512;

  for (int kt = 0; kt < ktiles; ++kt) {
    // ---- load phase: everything live before any WMMA ----
    const v16bf a0 = load_a_frag(arow0 + kt * 32, lda, lane);
    const v16bf a1 = load_a_frag(arow1 + kt * 32, lda, lane);
    const v16bf b0 = load_b_packed(wt0 + kt * 512, lane);
    const v16bf b1 = load_b_packed(wt1 + kt * 512, lane);
    const v16bf b2 = load_b_packed(wt2 + kt * 512, lane);
    const v16bf b3 = load_b_packed(wt3 + kt * 512, lane);
    if (kt + 1 < ktiles) __builtin_prefetch(wt0 + (kt + 1) * 512, 0, 3);
    // ---- math phase: 8 independent accumulator chains ----
    acc[0][0] = __builtin_amdgcn_wmma_f32_16x16x32_bf16(false, a0, false, b0, (short)0, acc[0][0], false, false);
    acc[1][0] = __builtin_amdgcn_wmma_f32_16x16x32_bf16(false, a1, false, b0, (short)0, acc[1][0], false, false);
    acc[0][1] = __builtin_amdgcn_wmma_f32_16x16x32_bf16(false, a0, false, b1, (short)0, acc[0][1], false, false);
    acc[1][1] = __builtin_amdgcn_wmma_f32_16x16x32_bf16(false, a1, false, b1, (short)0, acc[1][1], false, false);
    acc[0][2] = __builtin_amdgcn_wmma_f32_16x16x32_bf16(false, a0, false, b2, (short)0, acc[0][2], false, false);
    acc[1][2] = __builtin_amdgcn_wmma_f32_16x16x32_bf16(false, a1, false, b2, (short)0, acc[1][2], false, false);
    acc[0][3] = __builtin_amdgcn_wmma_f32_16x16x32_bf16(false, a0, false, b3, (short)0, acc[0][3], false, false);
    acc[1][3] = __builtin_amdgcn_wmma_f32_16x16x32_bf16(false, a1, false, b3, (short)0, acc[1][3], false, false);
  }

  const int n = lane & 15, mb = (lane >> 4) << 3;
#pragma unroll
  for (int r = 0; r < 2; ++r) {
#pragma unroll
    for (int t = 0; t < 4; ++t) {
      const int cn = col0 + t * 16 + n;
      const float bv = bias[cn];
#pragma unroll
      for (int v = 0; v < 8; ++v) {
        float val = acc[r][t][v] + bv;
        if (relu) val = fmaxf(val, 0.0f);
        C[(size_t)(row0 + r * 16 + mb + v) * ldc + cn] = val;
      }
    }
  }
  (void)N;
}

// ===========================================================================
// Flash-attention: Q,K,V laid out [B*H, S, kD] (f32). Output O in [B, S, H*kD].
// One wave per 16-row query tile; streams 16-key chunks with online softmax.
// ===========================================================================
__global__ __launch_bounds__(256) void attn_kernel(
    const float* __restrict__ Qg, const float* __restrict__ Kg,
    const float* __restrict__ Vg, float* __restrict__ O,
    int S_, int Hn, int causal)
{
  __shared__ float pbuf[8][16][32];
  const int lane = threadIdx.x & 31;
  const int wave = threadIdx.x >> 5;
  const int bh = blockIdx.y;
  const int b = bh / Hn, h = bh % Hn;
  const float* Qp = Qg + (size_t)bh * S_ * kD;
  const float* Kp = Kg + (size_t)bh * S_ * kD;
  const float* Vp = Vg + (size_t)bh * S_ * kD;
  const int qb = blockIdx.x * 128 + wave * 16;
  const int n  = lane & 15;
  const int mb = (lane >> 4) << 3;

  // zero this wave's P staging tile once; columns k=16..31 stay zero forever
  for (int i = lane; i < 16 * 32; i += 32) pbuf[wave][i >> 5][i & 31] = 0.0f;

  const v16bf aq0 = load_a_bf32(Qp + (size_t)qb * kD,      kD, lane);
  const v16bf aq1 = load_a_bf32(Qp + (size_t)qb * kD + 32, kD, lane);

  float rmax[8], rsum[8];
  v8f oacc[4];
#pragma unroll
  for (int v = 0; v < 8; ++v) { rmax[v] = -1e30f; rsum[v] = 0.0f; }
#pragma unroll
  for (int t = 0; t < 4; ++t) oacc[t] = zero8();

  const int kvEnd = causal ? (qb + 16) : S_;
  for (int kvb = 0; kvb < kvEnd; kvb += 16) {
    const v16bf bk0 = load_bT_bf(Kp, kvb, 0,  lane);
    const v16bf bk1 = load_bT_bf(Kp, kvb, 32, lane);
    v8f c = zero8();
    c = __builtin_amdgcn_wmma_f32_16x16x32_bf16(false, aq0, false, bk0, (short)0, c, false, false);
    c = __builtin_amdgcn_wmma_f32_16x16x32_bf16(false, aq1, false, bk1, (short)0, c, false, false);

#pragma unroll
    for (int v = 0; v < 8; ++v) {
      float s = c[v] * 0.125f;  // 1/sqrt(64)
      if (causal && (kvb + n) > (qb + mb + v)) s = -1e30f;
      float rm = s;
      rm = fmaxf(rm, __shfl_xor(rm, 1, 16));
      rm = fmaxf(rm, __shfl_xor(rm, 2, 16));
      rm = fmaxf(rm, __shfl_xor(rm, 4, 16));
      rm = fmaxf(rm, __shfl_xor(rm, 8, 16));
      const float mnew = fmaxf(rmax[v], rm);
      const float pv = __expf(s - mnew);
      float ps = pv;
      ps += __shfl_xor(ps, 1, 16);
      ps += __shfl_xor(ps, 2, 16);
      ps += __shfl_xor(ps, 4, 16);
      ps += __shfl_xor(ps, 8, 16);
      const float corr = __expf(rmax[v] - mnew);
      rsum[v] = rsum[v] * corr + ps;
      rmax[v] = mnew;
      pbuf[wave][mb + v][n] = pv;  // DS ops are in-order within a wave
#pragma unroll
      for (int t = 0; t < 4; ++t) oacc[t][v] *= corr;
    }

    const v16bf ap = load_ap_bf(pbuf[wave], lane);
#pragma unroll
    for (int t = 0; t < 4; ++t) {
      const v16bf bv = load_bv_bf(Vp, kvb, t * 16, S_, lane);
      oacc[t] = __builtin_amdgcn_wmma_f32_16x16x32_bf16(
          false, ap, false, bv, (short)0, oacc[t], false, false);
    }
  }

  float* Op = O + ((size_t)b * S_ + qb) * (Hn * kD) + h * kD;
#pragma unroll
  for (int t = 0; t < 4; ++t)
#pragma unroll
    for (int v = 0; v < 8; ++v)
      Op[(size_t)(mb + v) * (Hn * kD) + t * 16 + n] = oacc[t][v] / rsum[v];
}

// ===========================================================================
// Embedding gather + sinusoidal positional encoding
// ===========================================================================
__global__ __launch_bounds__(256) void embed_kernel(
    const int* __restrict__ tok, const float* __restrict__ emb, float* __restrict__ X)
{
  const int row = blockIdx.x;  // b*kS + s
  const int s = row & (kS - 1);
  const int t = tok[row];
  const float* er = emb + (size_t)t * kE;
  float* xr = X + (size_t)row * kE;
  for (int e = threadIdx.x; e < kE; e += 256) {
    const float i2  = (float)(e & ~1);
    const float ang = (float)s * __powf(10000.0f, -i2 / (float)kE);
    const float pe  = (e & 1) ? __cosf(ang) : __sinf(ang);
    xr[e] = er[e] + pe;
  }
}

// ===========================================================================
// Residual add + LayerNorm over E=1024 (one block per row)
// ===========================================================================
__global__ __launch_bounds__(256) void addnorm_kernel(
    const float* __restrict__ X, const float* __restrict__ Y,
    const float* __restrict__ g, const float* __restrict__ bta,
    float* __restrict__ O)
{
  __shared__ float redS[8], redQ[8];
  const size_t row = blockIdx.x;
  const float* x = X + row * kE;
  const float* y = Y + row * kE;
  const int t = threadIdx.x;
  float vals[kE / 256];
  float s = 0.0f, q = 0.0f;
#pragma unroll
  for (int i = 0; i < kE / 256; ++i) {
    const float v = x[t + i * 256] + y[t + i * 256];
    vals[i] = v; s += v; q += v * v;
  }
#pragma unroll
  for (int off = 16; off; off >>= 1) {
    s += __shfl_xor(s, off, 32);
    q += __shfl_xor(q, off, 32);
  }
  if ((t & 31) == 0) { redS[t >> 5] = s; redQ[t >> 5] = q; }
  __syncthreads();
  s = 0.0f; q = 0.0f;
#pragma unroll
  for (int i = 0; i < 8; ++i) { s += redS[i]; q += redQ[i]; }
  const float mean = s * (1.0f / kE);
  const float var  = q * (1.0f / kE) - mean * mean;
  const float inv  = rsqrtf(var + 1e-15f);
  float* o = O + row * kE;
#pragma unroll
  for (int i = 0; i < kE / 256; ++i) {
    const int e = t + i * 256;
    o[e] = g[e] * (vals[i] - mean) * inv + bta[e];
  }
}

// ===========================================================================
// In-place softmax over V=32000 (one block per row)
// ===========================================================================
__global__ __launch_bounds__(256) void softmax_kernel(float* __restrict__ L, int Vn)
{
  __shared__ float redM[8], redS[8];
  const size_t row = blockIdx.x;
  float* p = L + row * (size_t)Vn;
  const int t = threadIdx.x;
  float mx = -3.4e38f;
  for (int i = t; i < Vn; i += 256) mx = fmaxf(mx, p[i]);
#pragma unroll
  for (int off = 16; off; off >>= 1) mx = fmaxf(mx, __shfl_xor(mx, off, 32));
  if ((t & 31) == 0) redM[t >> 5] = mx;
  __syncthreads();
  float m8 = -3.4e38f;
#pragma unroll
  for (int i = 0; i < 8; ++i) m8 = fmaxf(m8, redM[i]);
  float s = 0.0f;
  for (int i = t; i < Vn; i += 256) s += __expf(p[i] - m8);
#pragma unroll
  for (int off = 16; off; off >>= 1) s += __shfl_xor(s, off, 32);
  if ((t & 31) == 0) redS[t >> 5] = s;
  __syncthreads();
  float tot = 0.0f;
#pragma unroll
  for (int i = 0; i < 8; ++i) tot += redS[i];
  const float inv = 1.0f / tot;
  for (int i = t; i < Vn; i += 256) p[i] = __expf(p[i] - m8) * inv;
}

// ===========================================================================
extern "C" void kernel_launch(void* const* d_in, const int* in_sizes, int n_in,
                              void* d_out, int out_size, void* d_ws, size_t ws_size,
                              hipStream_t stream) {
  const int*   src    = (const int*)  d_in[0];
  const int*   trg    = (const int*)  d_in[1];
  const float* emb_e  = (const float*)d_in[2];
  const float* emb_d  = (const float*)d_in[3];
  const float* qkv_e  = (const float*)d_in[4];
  const float* bqkv_e = (const float*)d_in[5];
  const float* wo_e   = (const float*)d_in[6];
  const float* bo_e   = (const float*)d_in[7];
  const float* qkv_m  = (const float*)d_in[8];
  const float* bqkv_m = (const float*)d_in[9];
  const float* wo_m   = (const float*)d_in[10];
  const float* bo_m   = (const float*)d_in[11];
  const float* qkv_c  = (const float*)d_in[12];
  const float* bqkv_c = (const float*)d_in[13];
  const float* wo_c   = (const float*)d_in[14];
  const float* bo_c   = (const float*)d_in[15];
  const float* gamma  = (const float*)d_in[16];
  const float* beta   = (const float*)d_in[17];
  const float* ff_w1  = (const float*)d_in[18];
  const float* ff_b1  = (const float*)d_in[19];
  const float* ff_w2  = (const float*)d_in[20];
  const float* ff_b2  = (const float*)d_in[21];
  const float* w_out  = (const float*)d_in[22];
  const float* b_out  = (const float*)d_in[23];
  float* out = (float*)d_out;

  char* ws = (char*)d_ws;
  const size_t BUFSZ = (size_t)kB * kS * kE;          // 2M elements
  float* bufX   = (float*)(ws + 0 * BUFSZ * 4);       // residual / n3d
  float* bufQ   = (float*)(ws + 1 * BUFSZ * 4);       // Q (also FF hidden)
  float* bufK   = (float*)(ws + 2 * BUFSZ * 4);
  float* bufV   = (float*)(ws + 3 * BUFSZ * 4);
  float* bufATT = (float*)(ws + 4 * BUFSZ * 4);       // attn out / FF out
  float* bufPRJ = (float*)(ws + 5 * BUFSZ * 4);
  float* bufN1  = (float*)(ws + 6 * BUFSZ * 4);       // norm1 / n2d
  float* bufENC = (float*)(ws + 7 * BUFSZ * 4);       // persists
  float* bufN1D = (float*)(ws + 8 * BUFSZ * 4);       // persists
  __bf16* bufAbf  = (__bf16*)(ws + 9 * BUFSZ * 4);            // 4 MB
  __bf16* bufWsm  = (__bf16*)(ws + 9 * BUFSZ * 4 + BUFSZ * 2); // 2 MB small packed W
  __bf16* bufWbig = (__bf16*)ws;  // packed w_out overlays dead f32 region at the end

  const int MROWS = kB * kS;  // 2048

  auto packA = [&](const float* in, int n) {
    pack_a_kernel<<<dim3(n / 1024), 256, 0, stream>>>(in, bufAbf, n);
  };
  auto packB = [&](const float* W, __bf16* dst, int K, int N, int Z,
                   long long inZ, long long outZ) {
    pack_b_kernel<<<dim3((K / 32) * (N / 16), 1, Z), 256, 0, stream>>>(W, dst, K, N, inZ, outZ);
  };
  // plain GEMM: packed A in bufAbf [M,K], packed W, f32 C [M,N]
  auto gemm = [&](const __bf16* Wp, const float* bias, float* C,
                  int M, int N, int K, int relu) {
    dim3 g(N / 64, M / 256, 1);
    gemm_bf_kernel<<<g, 256, 0, stream>>>(bufAbf, Wp, bias, C, M, N, K, K, N,
                                          0LL, 0LL, 0LL, 0LL, 1, relu);
  };
  // per-head QKV projection: packed X in bufAbf; qkvw comp [H,E,D] packed per head
  auto qkvproj = [&](const float* qkvw, const float* qkvb, int comp, float* dst) {
    packB(qkvw + (size_t)comp * kH * kE * kD, bufWsm, kE, kD, kH,
          (long long)kE * kD, (long long)kE * kD);
    dim3 g(kD / 64, kS / 256, kB * kH);
    gemm_bf_kernel<<<g, 256, 0, stream>>>(
        bufAbf, bufWsm, qkvb + (size_t)comp * kH * kD, dst,
        kS, kD, kE, kE, kD,
        (long long)kS * kE, (long long)kE * kD, (long long)kD, (long long)kS * kD,
        kH, 0);
  };
  auto attn = [&](const float* Q, const float* K, const float* V, float* O, int causal) {
    attn_kernel<<<dim3(kS / 128, kB * kH), 256, 0, stream>>>(Q, K, V, O, kS, kH, causal);
  };
  auto addnorm = [&](const float* X, const float* Y, int layer, float* O) {
    addnorm_kernel<<<dim3(MROWS), 256, 0, stream>>>(
        X, Y, gamma + (size_t)layer * kE, beta + (size_t)layer * kE, O);
  };

  // ---------------- encoder ----------------
  embed_kernel<<<dim3(MROWS), 256, 0, stream>>>(src, emb_e, bufX);
  packA(bufX, MROWS * kE);
  qkvproj(qkv_e, bqkv_e, 0, bufQ);
  qkvproj(qkv_e, bqkv_e, 1, bufK);
  qkvproj(qkv_e, bqkv_e, 2, bufV);
  attn(bufQ, bufK, bufV, bufATT, /*causal=*/0);
  packA(bufATT, MROWS * kE);
  packB(wo_e, bufWsm, kH * kD, kE, 1, 0LL, 0LL);
  gemm(bufWsm, bo_e, bufPRJ, MROWS, kE, kH * kD, 0);
  addnorm(bufX, bufPRJ, 0, bufN1);
  packA(bufN1, MROWS * kE);
  packB(ff_w1, bufWsm, kE, kF, 1, 0LL, 0LL);
  gemm(bufWsm, ff_b1, bufQ, MROWS, kF, kE, /*relu=*/1);
  packA(bufQ, MROWS * kF);
  packB(ff_w2, bufWsm, kF, kE, 1, 0LL, 0LL);
  gemm(bufWsm, ff_b2, bufATT, MROWS, kE, kF, 0);
  addnorm(bufN1, bufATT, 1, bufENC);

  // ---------------- decoder: masked self-attention ----------------
  embed_kernel<<<dim3(MROWS), 256, 0, stream>>>(trg, emb_d, bufX);
  packA(bufX, MROWS * kE);
  qkvproj(qkv_m, bqkv_m, 0, bufQ);
  qkvproj(qkv_m, bqkv_m, 1, bufK);
  qkvproj(qkv_m, bqkv_m, 2, bufV);
  attn(bufQ, bufK, bufV, bufATT, /*causal=*/1);
  packA(bufATT, MROWS * kE);
  packB(wo_m, bufWsm, kH * kD, kE, 1, 0LL, 0LL);
  gemm(bufWsm, bo_m, bufPRJ, MROWS, kE, kH * kD, 0);
  addnorm(bufX, bufPRJ, 2, bufN1D);

  // ---------------- decoder: cross attention ----------------
  packA(bufN1D, MROWS * kE);
  qkvproj(qkv_c, bqkv_c, 0, bufQ);
  packA(bufENC, MROWS * kE);
  qkvproj(qkv_c, bqkv_c, 1, bufK);
  qkvproj(qkv_c, bqkv_c, 2, bufV);
  attn(bufQ, bufK, bufV, bufATT, /*causal=*/0);
  packA(bufATT, MROWS * kE);
  packB(wo_c, bufWsm, kH * kD, kE, 1, 0LL, 0LL);
  gemm(bufWsm, bo_c, bufPRJ, MROWS, kE, kH * kD, 0);
  addnorm(bufN1D, bufPRJ, 3, bufN1);

  // ---------------- decoder FFN + final norm ----------------
  packA(bufN1, MROWS * kE);
  packB(ff_w1 + (size_t)kE * kF, bufWsm, kE, kF, 1, 0LL, 0LL);
  gemm(bufWsm, ff_b1 + kF, bufQ, MROWS, kF, kE, /*relu=*/1);
  packA(bufQ, MROWS * kF);
  packB(ff_w2 + (size_t)kF * kE, bufWsm, kF, kE, 1, 0LL, 0LL);
  gemm(bufWsm, ff_b2 + kE, bufATT, MROWS, kE, kF, 0);
  addnorm(bufN1, bufATT, 4, bufX);

  // ---------------- logits + softmax ----------------
  packA(bufX, MROWS * kE);                     // bufX consumed here ...
  packB(w_out, bufWbig, kE, kV, 1, 0LL, 0LL);  // ... then f32 region is overlaid
  gemm(bufWbig, b_out, out, MROWS, kV, kE, 0);
  softmax_kernel<<<dim3(MROWS), 256, 0, stream>>>(out, kV);

  (void)in_sizes; (void)n_in; (void)out_size; (void)ws_size;
}
